// DAGKT_38113539785080
// MI455X (gfx1250) — matrix operand order repaired and checked by reading, not compiled, not run on hardware
//
#include <hip/hip_runtime.h>
#include <cstdint>
#include <cstddef>

// Problem constants (match reference)
#define NN     60000
#define EE     600000
#define BBATCH 1024
#define LLAY   3
#define HHD    4
#define DDIM   64
#define FEAT   16
#define EFL    8

typedef __attribute__((ext_vector_type(16))) __bf16 v16bf;
typedef __attribute__((ext_vector_type(8)))  float  v8f;

// ---------------------------------------------------------------------------
// Pack a row-major fp32 weight matrix W[K][ncols] into the CDNA5 WMMA
// B-operand bf16 layout. B fragment (32x16 bf16, wave32): lane l holds
// column n = l&15, k-half g = l>>4; element e maps to k = g*8 + e (e<8),
// k = 16 + g*8 + (e-8) (e>=8).
// Buffer layout: [kt][nt][lane][16] bf16 so each lane's fragment is one
// contiguous 32-byte chunk.
// ---------------------------------------------------------------------------
__global__ __launch_bounds__(256) void pack_b(
    const float* __restrict__ W, int K, int ncols, int NT_total, int nt_off,
    __bf16* __restrict__ dst)
{
  int t   = blockIdx.x * 256 + threadIdx.x;
  int NTs = ncols >> 4;
  int KT  = (K + 31) >> 5;
  if (t >= KT * NTs * 512) return;
  int e    = t & 15;
  int lane = (t >> 4) & 31;
  int ntl  = (t >> 9) % NTs;
  int kt   = (t >> 9) / NTs;
  int g    = lane >> 4;
  int k    = kt * 32 + (e < 8 ? g * 8 + e : 16 + g * 8 + (e - 8));
  int n    = ntl * 16 + (lane & 15);
  float v  = (k < K) ? W[(size_t)k * ncols + n] : 0.f;
  dst[(((size_t)kt * NT_total + nt_off + ntl) * 32 + lane) * 16 + e] = (__bf16)v;
}

// ---------------------------------------------------------------------------
// WMMA GEMM: C[M x (NT_total*16)] = act(A[M x K] * B + bias)
// A fp32 row-major (stride lda), B pre-packed bf16, C fp32 (stride ldc).
// Block = 256 threads = 8 waves; each wave owns a 16x64 strip (4 wmma tiles),
// block tile = 128 x 64. act: 0=none, 1=ELU, 2=leaky(0.01).
// K is compile-time: K%32==0 -> branch-free float4 loads; K==8 -> uniform
// load of the 8 valid floats + lane select (no EXEC divergence, no OOB).
// ---------------------------------------------------------------------------
template <int K>
__global__ __launch_bounds__(256) void gemm_wmma(
    const float* __restrict__ A, int lda, int M,
    const __bf16* __restrict__ Bp, int NT_total,
    float* __restrict__ C, int ldc,
    const float* __restrict__ bias, int act)
{
  int wave = threadIdx.x >> 5;
  int lane = threadIdx.x & 31;
  int m16  = lane & 15;
  int g    = lane >> 4;
  long r0  = (long)blockIdx.x * 128 + wave * 16;
  int ntb  = blockIdx.y * 4;

  long arow = r0 + m16;
  if (arow >= M) arow = M - 1;           // clamp (EXEC must stay all-ones)
  const float* Ar = A + (size_t)arow * lda;

  v8f acc[4] = {};
  constexpr int KT = (K + 31) / 32;

  #pragma unroll
  for (int kt = 0; kt < KT; ++kt) {
    // A fragment 16x32 bf16: lane (m16,g); elems 0..7 -> k=kt*32+g*8+e,
    // elems 8..15 -> k=kt*32+16+g*8+(e-8): two contiguous 8-float runs.
    v16bf a;
    if constexpr (K % 32 == 0) {
      int base0 = kt * 32 + g * 8;
      float4 x0 = *(const float4*)(Ar + base0);
      float4 x1 = *(const float4*)(Ar + base0 + 4);
      float4 x2 = *(const float4*)(Ar + base0 + 16);
      float4 x3 = *(const float4*)(Ar + base0 + 20);
      a[0]=(__bf16)x0.x;  a[1]=(__bf16)x0.y;  a[2]=(__bf16)x0.z;  a[3]=(__bf16)x0.w;
      a[4]=(__bf16)x1.x;  a[5]=(__bf16)x1.y;  a[6]=(__bf16)x1.z;  a[7]=(__bf16)x1.w;
      a[8]=(__bf16)x2.x;  a[9]=(__bf16)x2.y;  a[10]=(__bf16)x2.z; a[11]=(__bf16)x2.w;
      a[12]=(__bf16)x3.x; a[13]=(__bf16)x3.y; a[14]=(__bf16)x3.z; a[15]=(__bf16)x3.w;
    } else {
      // K == 8 (zero-padded to one 32-step). Valid k are 0..7, owned by the
      // g==0 half (elems 0..7). All lanes load those 8 floats uniformly, the
      // g==1 half zeroes them; elems 8..15 map to k>=16 -> always zero.
      float4 x0 = *(const float4*)(Ar);
      float4 x1 = *(const float4*)(Ar + 4);
      float zm = (g == 0) ? 1.f : 0.f;
      a[0]=(__bf16)(x0.x*zm); a[1]=(__bf16)(x0.y*zm);
      a[2]=(__bf16)(x0.z*zm); a[3]=(__bf16)(x0.w*zm);
      a[4]=(__bf16)(x1.x*zm); a[5]=(__bf16)(x1.y*zm);
      a[6]=(__bf16)(x1.z*zm); a[7]=(__bf16)(x1.w*zm);
      #pragma unroll
      for (int e2 = 8; e2 < 16; ++e2) a[e2] = (__bf16)0.f;
    }
    const v16bf* bp =
        (const v16bf*)(Bp + (((size_t)kt * NT_total + ntb) * 32 + lane) * 16);
    #pragma unroll
    for (int t = 0; t < 4; ++t) {
      v16bf b = bp[t * 32];   // next n-tile is 512 bf16 = 32 fragments away
      acc[t] = __builtin_amdgcn_wmma_f32_16x16x32_bf16(
          false, a, false, b, (short)0, acc[t], false, false);
    }
  }

  // Epilogue. D layout: lane (n=m16, g); vgpr r -> row r+8g, col n.
  #pragma unroll
  for (int t = 0; t < 4; ++t) {
    int col  = (ntb + t) * 16 + m16;
    float bv = bias ? bias[col] : 0.f;
    #pragma unroll
    for (int r = 0; r < 8; ++r) {
      long row = r0 + r + 8 * g;
      if (row < M) {
        float v = acc[t][r] + bv;
        if (act == 1)      v = v > 0.f ? v : (__expf(v) - 1.f);
        else if (act == 2) v = v > 0.f ? v : 0.01f * v;
        C[(size_t)row * ldc + col] = v;
      }
    }
  }
}

// ---------------------------------------------------------------------------
// Order-preserving uint key for float atomic max
// ---------------------------------------------------------------------------
__device__ __forceinline__ unsigned fkey(float x) {
  unsigned b = __float_as_uint(x);
  return (b & 0x80000000u) ? ~b : (b | 0x80000000u);
}
__device__ __forceinline__ float funkey(unsigned k) {
  unsigned b = (k & 0x80000000u) ? (k ^ 0x80000000u) : ~k;
  return __uint_as_float(b);
}

// logits + segment max. xw: [N,384] = xWni | xWnj | h. efw: [E,64] (incl bias)
__global__ __launch_bounds__(256) void edge_logits(
    const float* __restrict__ xw, const float* __restrict__ efw,
    const int* __restrict__ src, const int* __restrict__ dst,
    const float* __restrict__ attn, float* __restrict__ elog,
    unsigned* __restrict__ mkey)
{
  int t = blockIdx.x * 256 + threadIdx.x;
  if (t >= EE * HHD) return;
  int h = t & 3;
  int e = t >> 2;
  int s = src[e], d = dst[e];
  const float* pi = xw + (size_t)s * 384 + h * FEAT;
  const float* pj = xw + (size_t)d * 384 + 64 + h * FEAT;
  const float* pe = efw + (size_t)e * 64 + h * FEAT;
  const float* at = attn + h * FEAT;
  float acc = 0.f;
  #pragma unroll
  for (int f = 0; f < FEAT; ++f) {
    float v = pi[f] + pj[f] + pe[f];
    v = v > 0.f ? v : 0.01f * v;          // leaky_relu(0.01)
    acc = fmaf(v, at[f], acc);
  }
  elog[t] = acc;
  atomicMax(&mkey[(size_t)d * HHD + h], fkey(acc));
}

// exp(e - max) + segment sum; overwrites elog with ex
__global__ __launch_bounds__(256) void edge_exp(
    float* __restrict__ elog, const int* __restrict__ dst,
    const unsigned* __restrict__ mkey, float* __restrict__ ssum)
{
  int t = blockIdx.x * 256 + threadIdx.x;
  if (t >= EE * HHD) return;
  int h = t & 3;
  int e = t >> 2;
  int d = dst[e];
  float m  = funkey(mkey[(size_t)d * HHD + h]);
  float ex = __expf(elog[t] - m);
  elog[t] = ex;
  atomicAdd(&ssum[(size_t)d * HHD + h], ex);
}

// out[dst] += h[src] * a ; one wave per edge, lane covers 8 contiguous cols
__global__ __launch_bounds__(256) void edge_agg(
    const float* __restrict__ xw, const int* __restrict__ src,
    const int* __restrict__ dst, const float* __restrict__ ex,
    const float* __restrict__ ssum, const float* __restrict__ mask,
    float* __restrict__ accb)
{
  int wid = (blockIdx.x * 256 + threadIdx.x) >> 5;
  if (wid >= EE) return;
  int lane = threadIdx.x & 31;
  int e = wid;
  int s = src[e], d = dst[e];
  int cb = lane * 8;                       // 0..248, [H=4][D=64] head-major
  int h  = cb >> 6;
  float a = ex[(size_t)e * HHD + h] / ssum[(size_t)d * HHD + h] * mask[e];
  const float* hp = xw + (size_t)s * 384 + 128 + cb;
  float* ap = accb + (size_t)d * 256 + cb;
  float4 v0 = *(const float4*)(hp);
  float4 v1 = *(const float4*)(hp + 4);
  atomicAdd(&ap[0], v0.x * a); atomicAdd(&ap[1], v0.y * a);
  atomicAdd(&ap[2], v0.z * a); atomicAdd(&ap[3], v0.w * a);
  atomicAdd(&ap[4], v1.x * a); atomicAdd(&ap[5], v1.y * a);
  atomicAdd(&ap[6], v1.z * a); atomicAdd(&ap[7], v1.w * a);
}

// ---------------------------------------------------------------------------
// Heads (B=1024, tiny)
// ---------------------------------------------------------------------------
__global__ __launch_bounds__(256) void head1_ui(
    const float* __restrict__ ui, const int* __restrict__ uidx,
    const int* __restrict__ iidx, const float* __restrict__ W,
    const float* __restrict__ b, float* __restrict__ hid)
{
  int t = blockIdx.x * 256 + threadIdx.x;
  if (t >= BBATCH * 128) return;
  int j = t & 127, r = t >> 7;
  const float* a0 = ui + (size_t)uidx[r] * (LLAY * DDIM);
  const float* a1 = ui + (size_t)iidx[r] * (LLAY * DDIM);
  float acc = b[j];
  for (int k = 0; k < LLAY * DDIM; ++k) acc = fmaf(a0[k], W[(size_t)k * 128 + j], acc);
  for (int k = 0; k < LLAY * DDIM; ++k)
    acc = fmaf(a1[k], W[(size_t)(LLAY * DDIM + k) * 128 + j], acc);
  hid[t] = fmaxf(acc, 0.f);
}

__global__ __launch_bounds__(256) void head1_subg(
    const float* __restrict__ sg, const int* __restrict__ sidx,
    const float* __restrict__ W, const float* __restrict__ b,
    float* __restrict__ hid)
{
  int t = blockIdx.x * 256 + threadIdx.x;
  if (t >= BBATCH * 128) return;
  int j = t & 127, r = t >> 7;
  const float* a0 = sg + (size_t)sidx[r] * (LLAY * DDIM);
  float acc = b[j];
  for (int k = 0; k < LLAY * DDIM; ++k) acc = fmaf(a0[k], W[(size_t)k * 128 + j], acc);
  hid[t] = fmaxf(acc, 0.f);
}

__global__ __launch_bounds__(256) void head2(
    const float* __restrict__ hid, const float* __restrict__ W2,
    const float* __restrict__ b2, float* __restrict__ out)
{
  int gt = blockIdx.x * 256 + threadIdx.x;
  int r = gt >> 5, lane = gt & 31;
  if (r >= BBATCH) return;
  float a = 0.f;
  for (int k = lane; k < 128; k += 32) a = fmaf(hid[(size_t)r * 128 + k], W2[k], a);
  #pragma unroll
  for (int off = 16; off > 0; off >>= 1) a += __shfl_down(a, off, 32);
  if (lane == 0) out[r] = 1.f / (1.f + __expf(-(a + b2[0])));
}

// ---------------------------------------------------------------------------
// One EGAT block (node gemm, edge gemm, segment softmax, scatter, agg gemm)
// KEDGE is the edge-feature inner dimension (8 local / 64 global).
// ---------------------------------------------------------------------------
template <int KEDGE>
static void egat_block(hipStream_t s, const float* x, int ldx,
                       const float* Wni, const float* Wnj, const float* Wn,
                       const float* Wfij, const float* ef,
                       const float* attn, const float* bias64,
                       const float* aggW, const float* aggB,
                       const float* mask, const int* src, const int* dst,
                       float* out, int ldo, int act,
                       float* xw, float* efw, float* elog,
                       unsigned* mkey, float* ssum, float* accb,
                       __bf16* pBn, __bf16* pBe, __bf16* pBa)
{
  // pack concatenated node weights: [Wni | Wnj | Wn] -> 24 n-tiles
  pack_b<<<16, 256, 0, s>>>(Wni, 64, 64, 24, 0, pBn);
  pack_b<<<16, 256, 0, s>>>(Wnj, 64, 64, 24, 4, pBn);
  pack_b<<<64, 256, 0, s>>>(Wn, 64, 256, 24, 8, pBn);
  dim3 g1((NN + 127) / 128, 6);
  gemm_wmma<64><<<g1, 256, 0, s>>>(x, ldx, NN, pBn, 24, xw, 384, nullptr, 0);

  // edge-feature gemm (+bias folded in)
  constexpr int KTe = (KEDGE + 31) / 32;
  pack_b<<<(KTe * 4 * 512 + 255) / 256, 256, 0, s>>>(Wfij, KEDGE, 64, 4, 0, pBe);
  dim3 g2((EE + 127) / 128, 1);
  gemm_wmma<KEDGE><<<g2, 256, 0, s>>>(ef, KEDGE, EE, pBe, 4, efw, 64, bias64, 0);

  hipMemsetAsync(mkey, 0, (size_t)NN * HHD * 4, s);   // key 0 < key(any float)
  hipMemsetAsync(ssum, 0, (size_t)NN * HHD * 4, s);
  edge_logits<<<(EE * HHD) / 256, 256, 0, s>>>(xw, efw, src, dst, attn, elog, mkey);
  edge_exp<<<(EE * HHD) / 256, 256, 0, s>>>(elog, dst, mkey, ssum);

  // accb aliases efw (efw dead after edge_logits) — zero after edge_exp
  hipMemsetAsync(accb, 0, (size_t)NN * 256 * 4, s);
  edge_agg<<<EE / 8, 256, 0, s>>>(xw, src, dst, elog, ssum, mask, accb);

  // aggregation gemm + bias + activation, into the per-layer concat slice
  pack_b<<<64, 256, 0, s>>>(aggW, 256, 64, 4, 0, pBa);
  dim3 g3((NN + 127) / 128, 1);
  gemm_wmma<256><<<g3, 256, 0, s>>>(accb, 256, NN, pBa, 4, out, ldo, aggB, act);
}

// ---------------------------------------------------------------------------
extern "C" void kernel_launch(void* const* d_in, const int* in_sizes, int n_in,
                              void* d_out, int out_size, void* d_ws, size_t ws_size,
                              hipStream_t stream)
{
  (void)in_sizes; (void)n_in; (void)out_size; (void)ws_size;
  const float* x0       = (const float*)d_in[0];
  const float* efeat    = (const float*)d_in[1];
  const float* efeat2   = (const float*)d_in[2];
  const float* mask     = (const float*)d_in[3];
  const float* mask2    = (const float*)d_in[4];
  const int*   src      = (const int*)d_in[5];
  const int*   dst      = (const int*)d_in[6];
  const int*   user_idx = (const int*)d_in[7];
  const int*   item_idx = (const int*)d_in[8];
  const int*   subg_idx = (const int*)d_in[9];
  const float* lW_node  = (const float*)d_in[10];
  const float* lW_ni    = (const float*)d_in[11];
  const float* lW_nj    = (const float*)d_in[12];
  const float* lW_fij   = (const float*)d_in[13];
  const float* l_attn   = (const float*)d_in[14];
  const float* l_bias   = (const float*)d_in[15];
  const float* agg1_W   = (const float*)d_in[16];
  const float* agg1_b   = (const float*)d_in[17];
  const float* gW_node  = (const float*)d_in[18];
  const float* gW_ni    = (const float*)d_in[19];
  const float* gW_nj    = (const float*)d_in[20];
  const float* gW_fij   = (const float*)d_in[21];
  const float* g_attn   = (const float*)d_in[22];
  const float* g_bias   = (const float*)d_in[23];
  const float* agg2_W   = (const float*)d_in[24];
  const float* agg2_b   = (const float*)d_in[25];
  const float* lin1_ui_W   = (const float*)d_in[26];
  const float* lin1_ui_b   = (const float*)d_in[27];
  const float* lin2_ui_W   = (const float*)d_in[28];
  const float* lin2_ui_b   = (const float*)d_in[29];
  const float* lin1_subg_W = (const float*)d_in[30];
  const float* lin1_subg_b = (const float*)d_in[31];
  const float* lin2_subg_W = (const float*)d_in[32];
  const float* lin2_subg_b = (const float*)d_in[33];

  // -------- workspace carve-up --------
  uintptr_t w = (uintptr_t)d_ws;
  auto carve = [&](size_t bytes) -> void* {
    void* p = (void*)w;
    w += (bytes + 255) & ~(size_t)255;
    return p;
  };
  float*    ui   = (float*)carve((size_t)NN * LLAY * DDIM * 4);  // 46 MB
  float*    sg   = (float*)carve((size_t)NN * LLAY * DDIM * 4);  // 46 MB
  float*    xw   = (float*)carve((size_t)NN * 384 * 4);          // 92 MB
  float*    efw  = (float*)carve((size_t)EE * 64 * 4);           // 154 MB
  float*    elog = (float*)carve((size_t)EE * HHD * 4);          // 9.6 MB
  unsigned* mkey = (unsigned*)carve((size_t)NN * HHD * 4);
  float*    ssum = (float*)carve((size_t)NN * HHD * 4);
  __bf16*   pBn  = (__bf16*)carve((size_t)2 * 24 * 512 * 2);
  __bf16*   pBe  = (__bf16*)carve((size_t)2 * 4 * 512 * 2);
  __bf16*   pBa  = (__bf16*)carve((size_t)8 * 4 * 512 * 2);
  float*    hid  = (float*)carve((size_t)BBATCH * 128 * 4);
  float*    accb = efw;  // alias: efw is dead by the time accb is written

  const float* xcur = x0;
  int ldx = DDIM;

  for (int i = 0; i < LLAY; ++i) {
    // local EGAT -> ELU -> ui[:, i*64 : i*64+64]
    egat_block<EFL>(stream, xcur, ldx,
               lW_ni + (size_t)i * 64 * 64, lW_nj + (size_t)i * 64 * 64,
               lW_node + (size_t)i * 64 * 256,
               lW_fij + (size_t)i * EFL * 64, efeat,
               l_attn + (size_t)i * HHD * FEAT, l_bias + (size_t)i * 64,
               agg1_W + (size_t)i * 256 * 64, agg1_b + (size_t)i * 64,
               mask, src, dst,
               ui + (size_t)i * DDIM, LLAY * DDIM, /*act=ELU*/1,
               xw, efw, elog, mkey, ssum, accb, pBn, pBe, pBa);
    xcur = ui + (size_t)i * DDIM; ldx = LLAY * DDIM;

    // global EGAT -> leaky -> sg[:, i*64 : i*64+64]
    egat_block<DDIM>(stream, xcur, ldx,
               gW_ni + (size_t)i * 64 * 64, gW_nj + (size_t)i * 64 * 64,
               gW_node + (size_t)i * 64 * 256,
               gW_fij + (size_t)i * 64 * 64, efeat2,
               g_attn + (size_t)i * HHD * FEAT, g_bias + (size_t)i * 64,
               agg2_W + (size_t)i * 256 * 64, agg2_b + (size_t)i * 64,
               mask2, src, dst,
               sg + (size_t)i * DDIM, LLAY * DDIM, /*act=leaky*/2,
               xw, efw, elog, mkey, ssum, accb, pBn, pBe, pBa);
    xcur = sg + (size_t)i * DDIM; ldx = LLAY * DDIM;
  }

  // -------- heads --------
  float* out = (float*)d_out;            // [0..B) subg, [B..2B) ui
  head1_subg<<<(BBATCH * 128) / 256, 256, 0, stream>>>(
      sg, subg_idx, lin1_subg_W, lin1_subg_b, hid);
  head2<<<(BBATCH * 32) / 256, 256, 0, stream>>>(
      hid, lin2_subg_W, lin2_subg_b, out);
  head1_ui<<<(BBATCH * 128) / 256, 256, 0, stream>>>(
      ui, user_idx, item_idx, lin1_ui_W, lin1_ui_b, hid);
  head2<<<(BBATCH * 32) / 256, 256, 0, stream>>>(
      hid, lin2_ui_W, lin2_ui_b, out + BBATCH);
}
